// Input_Embed_21534966022856
// MI455X (gfx1250) — compile-verified, matrix operand order checked by reading, not run on hardware
//
#include <hip/hip_runtime.h>
#include <hip/hip_fp16.h>

// ---------------------------------------------------------------------------
// PointNet++-style input embedding for MI455X (gfx1250, wave32).
//   FPS -> KNN -> [concat(center, nbr-center)] @ W^T -> BN -> hardswish
//       -> max over k -> transpose.
// Design notes (see analysis): all data L2-resident; BN stats computed
// analytically from 6x6 feat moments (27 floats) so the 134MB `y` tensor is
// never materialized; the conv GEMM uses V_WMMA_F32_16X16X4_F32 (K=6 padded
// to 8) with fused epilogue and max-over-k reduction in C-register layout.
// ---------------------------------------------------------------------------

#define BATCH   16
#define NPTS    4096
#define NPOINT  1024
#define KNN     32
#define EMBED   64
#define BN_EPS  1e-5f

typedef __attribute__((ext_vector_type(2)))  float    v2f;
typedef __attribute__((ext_vector_type(8)))  float    v8f;
typedef __attribute__((ext_vector_type(16))) _Float16 v16h;

#if __has_builtin(__builtin_amdgcn_wmma_f32_16x16x4_f32)
#define USE_WMMA_F32X4 1
#else
#define USE_WMMA_F32X4 0
#endif

// ---------------------------------------------------------------------------
// Kernel 1: farthest point sampling. One 1024-thread block per batch.
// Distance array lives in LDS; each of the 1024 serial steps does a
// 4-point/thread min-update followed by a block-wide argmax tree reduction.
// ---------------------------------------------------------------------------
__global__ __launch_bounds__(1024)
void fps_kernel(const float* __restrict__ xyz,
                float* __restrict__ new_xyz,
                int* __restrict__ fps_idx) {
    __shared__ float sDist[NPTS];
    __shared__ float sRv[1024];
    __shared__ int   sRi[1024];
    __shared__ int   sFar;
    __shared__ float sC[3];

    const int b = blockIdx.x;
    const int t = threadIdx.x;
    const float* bx = xyz + (size_t)b * NPTS * 3;

    for (int p = t; p < NPTS; p += 1024) sDist[p] = 1e10f;
    if (t == 0) sFar = 0;
    __syncthreads();

    for (int it = 0; it < NPOINT; ++it) {
        const int far = sFar;
        if (t == 0) {
            fps_idx[b * NPOINT + it] = far;
            sC[0] = bx[far * 3 + 0];
            sC[1] = bx[far * 3 + 1];
            sC[2] = bx[far * 3 + 2];
        }
        __syncthreads();
        const float cx = sC[0], cy = sC[1], cz = sC[2];
        if (t < 3) new_xyz[((size_t)b * NPOINT + it) * 3 + t] = sC[t];

        float bestV = -1.0f;
        int   bestI = 0;
        for (int p = t; p < NPTS; p += 1024) {
            const float dx = bx[p * 3 + 0] - cx;
            const float dy = bx[p * 3 + 1] - cy;
            const float dz = bx[p * 3 + 2] - cz;
            const float d  = dx * dx + dy * dy + dz * dz;
            const float nd = fminf(sDist[p], d);
            sDist[p] = nd;
            if (nd > bestV) { bestV = nd; bestI = p; }  // stride order keeps first-max
        }
        sRv[t] = bestV;
        sRi[t] = bestI;
        __syncthreads();
        for (int off = 512; off > 0; off >>= 1) {
            if (t < off) {
                const float ov = sRv[t + off];
                const int   oi = sRi[t + off];
                if (ov > sRv[t] || (ov == sRv[t] && oi < sRi[t])) {
                    sRv[t] = ov;
                    sRi[t] = oi;
                }
            }
            __syncthreads();
        }
        if (t == 0) sFar = sRi[0];
        __syncthreads();
    }
}

// ---------------------------------------------------------------------------
// Kernel 2: KNN (top-32 smallest squared distances, low-index tie-break).
// One wave32 per query; 4096-entry distance vector in LDS (2 waves/block ->
// 32KB). 32 rounds of wave argmin via shfl_xor butterfly.
// ---------------------------------------------------------------------------
#define KNN_WAVES 2
__global__ __launch_bounds__(KNN_WAVES * 32)
void knn_kernel(const float* __restrict__ xyz,
                const float* __restrict__ new_xyz,
                int* __restrict__ knn_idx) {
    __shared__ float sDist[KNN_WAVES][NPTS];

    const int wave = threadIdx.x >> 5;
    const int lane = threadIdx.x & 31;
    const int q = blockIdx.x * KNN_WAVES + wave;
    const int b = q >> 10;
    const float* bx = xyz + (size_t)b * NPTS * 3;

    const float cx = new_xyz[q * 3 + 0];
    const float cy = new_xyz[q * 3 + 1];
    const float cz = new_xyz[q * 3 + 2];

    float* dl = sDist[wave];
    for (int p = lane; p < NPTS; p += 32) {
        __builtin_prefetch(bx + (p + 256) * 3, 0, 0);  // global_prefetch_b8
        const float dx = bx[p * 3 + 0] - cx;
        const float dy = bx[p * 3 + 1] - cy;
        const float dz = bx[p * 3 + 2] - cz;
        dl[p] = dx * dx + dy * dy + dz * dz;
    }
    __syncthreads();

    for (int r = 0; r < KNN; ++r) {
        float v = 3.4e38f;
        int   iBest = NPTS;
        for (int p = lane; p < NPTS; p += 32) {
            const float d = dl[p];
            if (d < v) { v = d; iBest = p; }
        }
        for (int off = 16; off > 0; off >>= 1) {
            const float ov = __shfl_xor(v, off, 32);
            const int   oi = __shfl_xor(iBest, off, 32);
            if (ov < v || (ov == v && oi < iBest)) { v = ov; iBest = oi; }
        }
        if (lane == 0) {
            knn_idx[q * KNN + r] = iBest;
            dl[iBest] = 3.4e38f;
        }
        __syncthreads();
    }
}

// ---------------------------------------------------------------------------
// Kernel 3a/3b: feat first/second moments (6 sums + 21 upper-tri products).
// y = feat @ W^T  =>  mean_o = w_o . E[f];  E[y_o^2] = w_o^T E[f f^T] w_o.
// ---------------------------------------------------------------------------
__global__ void moments_zero_kernel(float* __restrict__ mom) {
    if (threadIdx.x < 27) mom[threadIdx.x] = 0.0f;
}

__global__ __launch_bounds__(256)
void moments_kernel(const float* __restrict__ xyz,
                    const float* __restrict__ new_xyz,
                    const int* __restrict__ knn_idx,
                    float* __restrict__ mom) {
    const int R = BATCH * NPOINT * KNN;
    float s1[6]  = {0.f, 0.f, 0.f, 0.f, 0.f, 0.f};
    float s2[21];
#pragma unroll
    for (int i = 0; i < 21; ++i) s2[i] = 0.f;

    for (int r = blockIdx.x * blockDim.x + threadIdx.x; r < R;
         r += gridDim.x * blockDim.x) {
        const int q = r >> 5;
        const int k = r & 31;
        const int b = q >> 10;
        float f[6];
        f[0] = new_xyz[q * 3 + 0];
        f[1] = new_xyz[q * 3 + 1];
        f[2] = new_xyz[q * 3 + 2];
        const int idx = knn_idx[q * KNN + k];
        const float* p = xyz + ((size_t)b * NPTS + idx) * 3;
        f[3] = p[0] - f[0];
        f[4] = p[1] - f[1];
        f[5] = p[2] - f[2];
        int u = 0;
#pragma unroll
        for (int c = 0; c < 6; ++c) {
            s1[c] += f[c];
#pragma unroll
            for (int d = c; d < 6; ++d) s2[u++] += f[c] * f[d];
        }
    }
    // wave32 reduction, then one atomicAdd burst per wave
    for (int off = 16; off > 0; off >>= 1) {
#pragma unroll
        for (int i = 0; i < 6; ++i)  s1[i] += __shfl_xor(s1[i], off, 32);
#pragma unroll
        for (int i = 0; i < 21; ++i) s2[i] += __shfl_xor(s2[i], off, 32);
    }
    if ((threadIdx.x & 31) == 0) {
#pragma unroll
        for (int i = 0; i < 6; ++i)  atomicAdd(&mom[i], s1[i]);
#pragma unroll
        for (int i = 0; i < 21; ++i) atomicAdd(&mom[6 + i], s2[i]);
    }
}

__global__ void bn_param_kernel(const float* __restrict__ conv_w,
                                const float* __restrict__ gamma,
                                const float* __restrict__ beta,
                                const float* __restrict__ mom,
                                float* __restrict__ scale,
                                float* __restrict__ bias) {
    const int o = threadIdx.x;
    if (o >= EMBED) return;
    const float inv_cnt = 1.0f / (float)(BATCH * NPOINT * KNN);
    float w[6];
#pragma unroll
    for (int c = 0; c < 6; ++c) w[c] = conv_w[o * 6 + c];
    float mean = 0.f;
#pragma unroll
    for (int c = 0; c < 6; ++c) mean += w[c] * mom[c];
    mean *= inv_cnt;
    float e2 = 0.f;
#pragma unroll
    for (int c = 0; c < 6; ++c) {
#pragma unroll
        for (int d = 0; d < 6; ++d) {
            const int u = (c <= d) ? c : d;
            const int v = (c <= d) ? d : c;
            const int idx = u * 6 - (u * (u + 1)) / 2 + v;  // upper-tri packing
            e2 += w[c] * w[d] * mom[6 + idx];
        }
    }
    e2 *= inv_cnt;
    const float var = e2 - mean * mean;
    const float sc = gamma[o] * rsqrtf(var + BN_EPS);
    scale[o] = sc;
    bias[o]  = beta[o] - mean * sc;
}

// ---------------------------------------------------------------------------
// Kernel 4: fused feat-build -> WMMA GEMM (K=6) -> BN affine -> hardswish
//           -> max over k=32 -> transposed store (b, ch, s).
// One wave per (query, 16-channel tile). Per ISA C-layout: VGPR j holds
// rows M=j (lanes 0-15) / M=j+8 (lanes 16-31), so max-over-16-rows is an
// in-register max + shfl_xor(16).
// ---------------------------------------------------------------------------
__global__ __launch_bounds__(256)
void embed_wmma_kernel(const float* __restrict__ xyz,
                       const float* __restrict__ new_xyz,
                       const int* __restrict__ knn_idx,
                       const float* __restrict__ conv_w,
                       const float* __restrict__ scale,
                       const float* __restrict__ bias,
                       float* __restrict__ out_x) {
    const int lane  = threadIdx.x & 31;
    const int wid   = blockIdx.x * 8 + (threadIdx.x >> 5);
    const int ntile = wid & 3;
    const int q     = wid >> 2;
    const int b     = q >> 10;
    const int s     = q & 1023;
    const int mrow  = lane & 15;
    const int half  = lane >> 4;
    const int ch    = ntile * 16 + (lane & 15);

    const float cx = new_xyz[q * 3 + 0];
    const float cy = new_xyz[q * 3 + 1];
    const float cz = new_xyz[q * 3 + 2];
    const float sc = scale[ch];
    const float bi = bias[ch];

    // B operand: W^T tile, B[kk][n] = conv_w[ch][kk], kk >= 6 padded with 0.
    const float* wrow = conv_w + ch * 6;
#if USE_WMMA_F32X4
    // 32-bit operand layout (16x16x4): VGPR0 = K(half*2), VGPR1 = K(half*2+1)
    v2f bop0, bop1;
    bop0.x = wrow[half * 2 + 0];
    bop0.y = wrow[half * 2 + 1];
    bop1.x = half ? 0.f : wrow[4];
    bop1.y = half ? 0.f : wrow[5];
#else
    // f16 fallback: B 32x16, lanes 0-15 hold K=0..7,16..23, lanes 16-31 K=8..15,24..31.
    v16h bh = {};
#pragma unroll
    for (int kk = 0; kk < 6; ++kk)
        bh[kk] = half ? (_Float16)0 : (_Float16)wrow[kk];
#endif

    float mtmax[2];
#pragma unroll
    for (int mt = 0; mt < 2; ++mt) {
        const int k   = mt * 16 + mrow;
        const int idx = knn_idx[q * KNN + k];
        const float* p = xyz + ((size_t)b * NPTS + idx) * 3;
        const float f0 = cx, f1 = cy, f2 = cz;
        const float f3 = p[0] - cx;
        const float f4 = p[1] - cy;
        const float f5 = p[2] - cz;

        v8f acc = {};
#if USE_WMMA_F32X4
        // A 16x4 layout: lanes 0-15 hold K=0,1; lanes 16-31 hold K=2,3.
        v2f a0, a1;
        a0.x = half ? f2 : f0;
        a0.y = half ? f3 : f1;
        a1.x = half ? 0.f : f4;
        a1.y = half ? 0.f : f5;
        acc = __builtin_amdgcn_wmma_f32_16x16x4_f32(false, a0, false, bop0,
                                                    (short)0, acc, false, false);
        acc = __builtin_amdgcn_wmma_f32_16x16x4_f32(false, a1, false, bop1,
                                                    (short)0, acc, false, false);
#else
        // A 16x32 f16: lanes 0-15 h[0..7]=K0..7, h[8..15]=K16..23 (all pad 0);
        // lanes 16-31 cover K8..15 / K24..31 -> all zero here.
        v16h ah = {};
        ah[0] = half ? (_Float16)0 : (_Float16)f0;
        ah[1] = half ? (_Float16)0 : (_Float16)f1;
        ah[2] = half ? (_Float16)0 : (_Float16)f2;
        ah[3] = half ? (_Float16)0 : (_Float16)f3;
        ah[4] = half ? (_Float16)0 : (_Float16)f4;
        ah[5] = half ? (_Float16)0 : (_Float16)f5;
        acc = __builtin_amdgcn_wmma_f32_16x16x32_f16(false, ah, false, bh,
                                                     (short)0, acc, false, false);
#endif
        // Fused epilogue: BN affine + hardswish, then max over this tile's
        // 16 k-rows (8 in-register + cross-half via shfl_xor 16).
        float m = -3.4e38f;
#pragma unroll
        for (int j = 0; j < 8; ++j) {
            const float y = acc[j] * sc + bi;
            const float h = y * fminf(fmaxf(y + 3.f, 0.f), 6.f) * (1.f / 6.f);
            m = fmaxf(m, h);
        }
        m = fmaxf(m, __shfl_xor(m, 16, 32));
        mtmax[mt] = m;
    }

    if (lane < 16)
        out_x[((size_t)b * EMBED + ch) * NPOINT + s] = fmaxf(mtmax[0], mtmax[1]);
}

// ---------------------------------------------------------------------------
// Host launcher.
// Inputs: xyz (16,4096,3) f32; conv_w (64,6) f32; bn_gamma (64); bn_beta (64).
// Output: [ new_xyz (16,1024,3) | x (16,64,1024) ] f32, concatenated flat.
// Workspace: fps_idx (64KB) + knn_idx (2MB) + moments/scale/bias (~1KB).
// ---------------------------------------------------------------------------
extern "C" void kernel_launch(void* const* d_in, const int* in_sizes, int n_in,
                              void* d_out, int out_size, void* d_ws, size_t ws_size,
                              hipStream_t stream) {
    const float* xyz   = (const float*)d_in[0];
    const float* convw = (const float*)d_in[1];
    const float* gamma = (const float*)d_in[2];
    const float* beta  = (const float*)d_in[3];

    float* out     = (float*)d_out;
    float* new_xyz = out;                          // B*NPOINT*3
    float* out_x   = out + BATCH * NPOINT * 3;     // B*EMBED*NPOINT

    char* ws = (char*)d_ws;
    int*   fps_idx = (int*)ws;
    int*   knn_idx = (int*)(ws + (size_t)BATCH * NPOINT * sizeof(int));
    float* mom     = (float*)(ws + (size_t)BATCH * NPOINT * sizeof(int)
                                 + (size_t)BATCH * NPOINT * KNN * sizeof(int));
    float* scale   = mom + 32;
    float* bias    = scale + EMBED;

    fps_kernel<<<BATCH, 1024, 0, stream>>>(xyz, new_xyz, fps_idx);
    knn_kernel<<<(BATCH * NPOINT) / KNN_WAVES, KNN_WAVES * 32, 0, stream>>>(
        xyz, new_xyz, knn_idx);
    moments_zero_kernel<<<1, 64, 0, stream>>>(mom);
    moments_kernel<<<1024, 256, 0, stream>>>(xyz, new_xyz, knn_idx, mom);
    bn_param_kernel<<<1, EMBED, 0, stream>>>(convw, gamma, beta, mom, scale, bias);
    embed_wmma_kernel<<<(BATCH * NPOINT * 4) / 8, 256, 0, stream>>>(
        xyz, new_xyz, knn_idx, convw, scale, bias, out_x);
}
